// Encode_62457414418590
// MI455X (gfx1250) — matrix operand order repaired and checked by reading, not compiled, not run on hardware
//
#include <hip/hip_runtime.h>
#include <hip/hip_bf16.h>

typedef _Float16 half_t;
typedef __attribute__((ext_vector_type(16))) _Float16 v16h;
typedef __attribute__((ext_vector_type(8)))  _Float16 v8h;
typedef __attribute__((ext_vector_type(8)))  float    v8f;

__device__ __forceinline__ float gelu_f(float x) {
    return 0.5f * x * (1.0f + erff(x * 0.7071067811865476f));
}

// ---------------------------------------------------------------------------
// WMMA GEMM:  out[m][n] = act( sum_k A[m][k]*W[n][k] + bias[n] ) + resid[m][n]
// A: (M x K) f16 row-major.  W: (N x K) f16 row-major (i.e. torch-style weight).
// One wave per 16x16 output tile, K stepped by 32 with v_wmma_f32_16x16x32_f16.
// ---------------------------------------------------------------------------
__global__ __launch_bounds__(256)
void k_gemm_wmma(const half_t* __restrict__ A, const half_t* __restrict__ W,
                 const float* __restrict__ bias, const float* __restrict__ resid,
                 float* __restrict__ outF, half_t* __restrict__ outH,
                 int M, int N, int K, int act)
{
    int wid  = (blockIdx.x * blockDim.x + threadIdx.x) >> 5;
    int lane = threadIdx.x & 31;
    int tilesN = N >> 4;
    int nTiles = (M >> 4) * tilesN;
    if (wid >= nTiles) return;
    int tm = (wid / tilesN) << 4;
    int tn = (wid % tilesN) << 4;
    int l  = lane & 15;
    int hf = lane >> 4;

    const half_t* ap = A + (size_t)(tm + l) * K + hf * 8;   // A frag: k = kk+hf*8+e (e<8), +16 for e>=8
    const half_t* bp = W + (size_t)(tn + l) * K + hf * 8;   // B frag: k = kk+hf*8+e (e<8), +16 for e>=8

    v8f c = {};
    for (int kk = 0; kk < K; kk += 32) {
        v8h a0 = *(const v8h*)(ap + kk);
        v8h a1 = *(const v8h*)(ap + kk + 16);
        v16h a = __builtin_shufflevector(a0, a1, 0,1,2,3,4,5,6,7,8,9,10,11,12,13,14,15);
        v8h b0 = *(const v8h*)(bp + kk);
        v8h b1 = *(const v8h*)(bp + kk + 16);
        v16h b = __builtin_shufflevector(b0, b1, 0,1,2,3,4,5,6,7,8,9,10,11,12,13,14,15);
        if (kk + 512 < K) {
            __builtin_prefetch(ap + kk + 512, 0, 1);
            __builtin_prefetch(bp + kk + 512, 0, 1);
        }
        c = __builtin_amdgcn_wmma_f32_16x16x32_f16(false, a, false, b, (short)0, c, false, false);
    }
#pragma unroll
    for (int r = 0; r < 8; r++) {
        int m = tm + hf * 8 + r;       // C/D layout: VGPR r, lanes0-15 M=r, lanes16-31 M=8+r
        int n = tn + l;
        size_t o = (size_t)m * N + n;
        float v = c[r];
        if (bias)  v += bias[n];
        if (act == 1) v = gelu_f(v);
        if (resid) v += resid[o];
        if (outF) outF[o] = v;
        if (outH) outH[o] = (half_t)v;
    }
}

// f32 -> f16 convert
__global__ void k_cvt(const float* __restrict__ in, half_t* __restrict__ out, int n)
{
    int i = blockIdx.x * blockDim.x + threadIdx.x;
    if (i < n) out[i] = (half_t)in[i];
}

// LayerNorm over rows of length C; optional f32 and f16 outputs
__global__ void k_ln(const float* __restrict__ in, const float* __restrict__ g,
                     const float* __restrict__ be, float* __restrict__ outF,
                     half_t* __restrict__ outH, int C)
{
    int row = blockIdx.x, tid = threadIdx.x;
    __shared__ float red[256];
    __shared__ float mv[2];
    const float* p = in + (size_t)row * C;
    float s = 0.f;
    for (int i = tid; i < C; i += 256) s += p[i];
    red[tid] = s; __syncthreads();
    for (int o = 128; o > 0; o >>= 1) { if (tid < o) red[tid] += red[tid + o]; __syncthreads(); }
    if (tid == 0) mv[0] = red[0] / (float)C;
    __syncthreads();
    float m = mv[0], ss = 0.f;
    for (int i = tid; i < C; i += 256) { float d = p[i] - m; ss += d * d; }
    red[tid] = ss; __syncthreads();
    for (int o = 128; o > 0; o >>= 1) { if (tid < o) red[tid] += red[tid + o]; __syncthreads(); }
    if (tid == 0) mv[1] = rsqrtf(red[0] / (float)C + 1e-5f);
    __syncthreads();
    float r = mv[1];
    for (int i = tid; i < C; i += 256) {
        float v = (p[i] - m) * r * g[i] + be[i];
        if (outF) outF[(size_t)row * C + i] = v;
        if (outH) outH[(size_t)row * C + i] = (half_t)v;
    }
}

// f = w_it @ x + b_it  : (4,32,2048)
__global__ void k_intrans(const float* __restrict__ x, const float* __restrict__ w,
                          const float* __restrict__ bias, float* __restrict__ f)
{
    int i = blockIdx.x * blockDim.x + threadIdx.x;
    if (i >= 4 * 32 * 2048) return;
    int n = i & 2047, oc = (i >> 11) & 31, b = i >> 16;
    float acc = bias[oc];
    for (int c = 0; c < 3; c++) acc += w[oc * 3 + c] * x[((size_t)(b * 3 + c)) * 2048 + n];
    f[i] = acc;
}

// 16-NN in coordinate space (self included), ascending distance, stable ties
__global__ void k_knn(const float* __restrict__ x, int* __restrict__ idx)
{
    int i = blockIdx.x * blockDim.x + threadIdx.x;
    if (i >= 4 * 2048) return;
    int b = i >> 11, n = i & 2047;
    const float* xb = x + (size_t)b * 3 * 2048;
    float px = xb[n], py = xb[2048 + n], pz = xb[4096 + n];
    float bd[16]; int bi[16];
    for (int t = 0; t < 16; t++) { bd[t] = 1e30f; bi[t] = 0; }
    for (int j = 0; j < 2048; j++) {
        float dx = xb[j] - px, dy = xb[2048 + j] - py, dz = xb[4096 + j] - pz;
        float d = dx * dx + dy * dy + dz * dz;
        if (d < bd[15]) {
            int p = 15;
            while (p > 0 && bd[p - 1] > d) { bd[p] = bd[p - 1]; bi[p] = bi[p - 1]; p--; }
            bd[p] = d; bi[p] = j;
        }
    }
    for (int t = 0; t < 16; t++) idx[(size_t)i * 16 + t] = bi[t];
}

// EdgeConv chain with 3-pass GroupNorm (recompute; deterministic partial sums)
// stage 0: conv1, emit gn1 per-(b,grp) partial sums
// stage 1: conv1 -> gn1 -> leaky -> conv2, emit gn2 partial sums
// stage 2: full chain -> max over K -> knn_f (B,128,N)
__global__ __launch_bounds__(256)
void k_edgeconv(const float* __restrict__ f, const int* __restrict__ idx,
                const float* __restrict__ wl1, const float* __restrict__ wl2,
                const float* __restrict__ gn1g, const float* __restrict__ gn1b,
                const float* __restrict__ gn2g, const float* __restrict__ gn2b,
                const float* __restrict__ mr, float* __restrict__ part,
                float* __restrict__ knn_out, int stage)
{
    int b = blockIdx.x >> 11, n = blockIdx.x & 2047;
    int tid = threadIdx.x;
    __shared__ float g[16 * 64];
    __shared__ float h1s[256 * 16];
    __shared__ float h2s[128 * 16];
    __shared__ float red[256];

    for (int t = tid; t < 16 * 64; t += 256) {
        int kk = t >> 6, c = t & 63, cm = c & 31;
        float center = f[((size_t)(b * 32 + cm)) * 2048 + n];
        float v;
        if (c < 32) {
            int nb = idx[((size_t)(b * 2048 + n)) * 16 + kk];
            v = f[((size_t)(b * 32 + cm)) * 2048 + nb] - center;
        } else v = center;
        g[kk * 64 + c] = v;
    }
    __syncthreads();
    for (int t = tid; t < 4096; t += 256) {          // conv1: 256 oc x 16 k
        int oc = t >> 4, kk = t & 15;
        const float* w = wl1 + oc * 64;
        const float* gg = g + kk * 64;
        float acc = 0.f;
        for (int c = 0; c < 64; c++) acc += w[c] * gg[c];
        h1s[t] = acc;
    }
    __syncthreads();
    if (stage == 0) {
        for (int grp = 0; grp < 4; grp++) {
            float s = 0.f, ss = 0.f;
            for (int t = tid; t < 1024; t += 256) { float v = h1s[grp * 1024 + t]; s += v; ss += v * v; }
            red[tid] = s; __syncthreads();
            for (int o = 128; o > 0; o >>= 1) { if (tid < o) red[tid] += red[tid + o]; __syncthreads(); }
            if (tid == 0) part[((size_t)((b * 4 + grp) * 2 + 0)) * 2048 + n] = red[0];
            __syncthreads();
            red[tid] = ss; __syncthreads();
            for (int o = 128; o > 0; o >>= 1) { if (tid < o) red[tid] += red[tid + o]; __syncthreads(); }
            if (tid == 0) part[((size_t)((b * 4 + grp) * 2 + 1)) * 2048 + n] = red[0];
            __syncthreads();
        }
        return;
    }
    for (int t = tid; t < 4096; t += 256) {          // gn1 + leaky
        int oc = t >> 4, grp = oc >> 6;
        float m = mr[(b * 4 + grp) * 2], r = mr[(b * 4 + grp) * 2 + 1];
        float v = (h1s[t] - m) * r * gn1g[oc] + gn1b[oc];
        h1s[t] = v >= 0.f ? v : 0.2f * v;
    }
    __syncthreads();
    for (int t = tid; t < 2048; t += 256) {          // conv2: 128 oc x 16 k
        int oc = t >> 4, kk = t & 15;
        const float* w = wl2 + oc * 256;
        float acc = 0.f;
        for (int c = 0; c < 256; c++) acc += w[c] * h1s[c * 16 + kk];
        h2s[t] = acc;
    }
    __syncthreads();
    if (stage == 1) {
        for (int grp = 0; grp < 4; grp++) {
            float s = 0.f, ss = 0.f;
            for (int t = tid; t < 512; t += 256) { float v = h2s[grp * 512 + t]; s += v; ss += v * v; }
            red[tid] = s; __syncthreads();
            for (int o = 128; o > 0; o >>= 1) { if (tid < o) red[tid] += red[tid + o]; __syncthreads(); }
            if (tid == 0) part[((size_t)((b * 4 + grp) * 2 + 0)) * 2048 + n] = red[0];
            __syncthreads();
            red[tid] = ss; __syncthreads();
            for (int o = 128; o > 0; o >>= 1) { if (tid < o) red[tid] += red[tid + o]; __syncthreads(); }
            if (tid == 0) part[((size_t)((b * 4 + grp) * 2 + 1)) * 2048 + n] = red[0];
            __syncthreads();
        }
        return;
    }
    for (int t = tid; t < 2048; t += 256) {          // gn2 + leaky
        int oc = t >> 4, grp = oc >> 5;
        float m = mr[32 + (b * 4 + grp) * 2], r = mr[32 + (b * 4 + grp) * 2 + 1];
        float v = (h2s[t] - m) * r * gn2g[oc] + gn2b[oc];
        h2s[t] = v >= 0.f ? v : 0.2f * v;
    }
    __syncthreads();
    if (tid < 128) {                                  // max over the 16 neighbors
        float mx = -1e30f;
        for (int kk = 0; kk < 16; kk++) mx = fmaxf(mx, h2s[tid * 16 + kk]);
        knn_out[((size_t)(b * 128 + tid)) * 2048 + n] = mx;
    }
}

// deterministic reduction of per-point partial sums -> stats[base+row]
__global__ void k_redstats(const float* __restrict__ part, float* __restrict__ stats, int base)
{
    int row = blockIdx.x, tid = threadIdx.x;
    __shared__ float red[256];
    float s = 0.f;
    for (int t = tid; t < 2048; t += 256) s += part[(size_t)row * 2048 + t];
    red[tid] = s; __syncthreads();
    for (int o = 128; o > 0; o >>= 1) { if (tid < o) red[tid] += red[tid + o]; __syncthreads(); }
    if (tid == 0) stats[base + row] = red[0];
}

__global__ void k_gnfin(const float* __restrict__ stats, float* __restrict__ mr, int which)
{
    int i = threadIdx.x;
    if (i >= 16) return;
    float cnt = (which == 0) ? (64.f * 2048.f * 16.f) : (32.f * 2048.f * 16.f);
    int base = (which == 0) ? 0 : 32;
    float m = stats[base + 2 * i] / cnt;
    float v = stats[base + 2 * i + 1] / cnt - m * m;
    mr[base + 2 * i]     = m;
    mr[base + 2 * i + 1] = rsqrtf(v + 1e-5f);
}

// scalar GEMM (tiny M): out[m][n] = sum_k A[m][k]*W[n][k] + bias
__global__ void k_gemm_scalar(const float* __restrict__ A, const float* __restrict__ W,
                              const float* __restrict__ bias, float* __restrict__ out,
                              int M, int N, int K)
{
    int i = blockIdx.x * blockDim.x + threadIdx.x;
    if (i >= M * N) return;
    int m = i / N, n = i % N;
    const float* a = A + (size_t)m * K;
    const float* w = W + (size_t)n * K;
    float acc = 0.f;
    for (int k = 0; k < K; k++) acc += a[k] * w[k];
    if (bias) acc += bias[n];
    out[i] = acc;
}

// 3-token attention (e1), flg=True: out = nx - attn(nx)
__global__ void k_attn1(const float* __restrict__ nx, const float* __restrict__ qkv,
                        float* __restrict__ out)
{
    int b = blockIdx.x >> 3, h = blockIdx.x & 7, d = threadIdx.x;  // 256 threads
    __shared__ float red[256];
    __shared__ float s[9];
    float q[3], k[3], v[3];
    for (int i = 0; i < 3; i++) {
        size_t base = ((size_t)(b * 3 + i)) * 6144 + h * 256 + d;
        q[i] = qkv[base]; k[i] = qkv[base + 2048]; v[i] = qkv[base + 4096];
    }
    for (int i = 0; i < 3; i++)
        for (int j = 0; j < 3; j++) {
            red[d] = q[i] * k[j];
            __syncthreads();
            for (int o = 128; o > 0; o >>= 1) { if (d < o) red[d] += red[d + o]; __syncthreads(); }
            if (d == 0) s[i * 3 + j] = red[0] * 0.0625f;   // hd^-0.5 = 1/16
            __syncthreads();
        }
    for (int i = 0; i < 3; i++) {
        float m = fmaxf(s[i * 3], fmaxf(s[i * 3 + 1], s[i * 3 + 2]));
        float e0 = expf(s[i * 3] - m), e1 = expf(s[i * 3 + 1] - m), e2 = expf(s[i * 3 + 2] - m);
        float inv = 1.f / (e0 + e1 + e2);
        float o = (e0 * v[0] + e1 * v[1] + e2 * v[2]) * inv;
        size_t oi = ((size_t)(b * 3 + i)) * 2048 + h * 256 + d;
        out[oi] = nx[oi] - o;
    }
}

// h1a = gelu(e1_c1 @ x1) : (4,512,2048)
__global__ void k_h1a(const float* __restrict__ x1, const float* __restrict__ c1,
                      float* __restrict__ out)
{
    int i = blockIdx.x * blockDim.x + threadIdx.x;
    if (i >= 4 * 512 * 2048) return;
    int n = i & 2047, oc = (i >> 11) & 511, b = i >> 20;
    float acc = c1[oc * 3 + 0] * x1[((size_t)(b * 3 + 0)) * 2048 + n]
              + c1[oc * 3 + 1] * x1[((size_t)(b * 3 + 1)) * 2048 + n]
              + c1[oc * 3 + 2] * x1[((size_t)(b * 3 + 2)) * 2048 + n];
    out[i] = gelu_f(acc);
}

// h1 = e1_c2 @ h1a : rows (b*128+oc) x 2048
__global__ void k_h1(const float* __restrict__ h1a, const float* __restrict__ c2,
                     float* __restrict__ out)
{
    int i = blockIdx.x * blockDim.x + threadIdx.x;
    if (i >= 4 * 128 * 2048) return;
    int n = i & 2047, oc = (i >> 11) & 127, b = i >> 18;
    const float* w  = c2 + oc * 512;
    const float* hb = h1a + (size_t)b * 512 * 2048 + n;
    float acc = 0.f;
    for (int c = 0; c < 512; c++) acc += w[c] * hb[(size_t)c * 2048];
    out[i] = acc;
}

// A_merged f16 (512,4096) = [h1 | knn_f]
__global__ void k_pack(const float* __restrict__ h1, const float* __restrict__ knn,
                       half_t* __restrict__ out)
{
    int i = blockIdx.x * blockDim.x + threadIdx.x;
    if (i >= 512 * 4096) return;
    int m = i >> 12, c = i & 4095;
    float v = (c < 2048) ? h1[(size_t)m * 2048 + c] : knn[(size_t)m * 2048 + (c - 2048)];
    out[i] = (half_t)v;
}

// attn2 scores: S (4,8,128,128) = (1/16) Q K^T
__global__ void k_scores(const float* __restrict__ qkv, float* __restrict__ S)
{
    int i = blockIdx.x * blockDim.x + threadIdx.x;
    if (i >= 4 * 8 * 128 * 128) return;
    int j = i & 127, ii = (i >> 7) & 127, h = (i >> 14) & 7, b = i >> 17;
    const float* q = qkv + ((size_t)(b * 128 + ii)) * 6144 + h * 256;
    const float* k = qkv + ((size_t)(b * 128 + j))  * 6144 + 2048 + h * 256;
    float acc = 0.f;
    for (int d = 0; d < 256; d++) acc += q[d] * k[d];
    S[i] = acc * 0.0625f;
}

__global__ void k_softmax128(float* __restrict__ S)
{
    int row = blockIdx.x, tid = threadIdx.x;   // 128 threads
    __shared__ float red[128];
    float v = S[(size_t)row * 128 + tid];
    red[tid] = v; __syncthreads();
    for (int o = 64; o > 0; o >>= 1) { if (tid < o) red[tid] = fmaxf(red[tid], red[tid + o]); __syncthreads(); }
    float m = red[0]; __syncthreads();
    float e = expf(v - m);
    red[tid] = e; __syncthreads();
    for (int o = 64; o > 0; o >>= 1) { if (tid < o) red[tid] += red[tid + o]; __syncthreads(); }
    S[(size_t)row * 128 + tid] = e / red[0];
}

// o = S @ V, write f32 + f16
// i = b*2^18 | h*2^15 | ii*2^8 | d   (d:0-7, ii:8-14, h:15-17, b:18-19)
__global__ void k_attnO(const float* __restrict__ S, const float* __restrict__ qkv,
                        float* __restrict__ outF, half_t* __restrict__ outH)
{
    int i = blockIdx.x * blockDim.x + threadIdx.x;
    if (i >= 4 * 8 * 128 * 256) return;
    int d = i & 255, ii = (i >> 8) & 127, h = (i >> 15) & 7, b = i >> 18;
    const float* srow = S + ((size_t)((b * 8 + h) * 128 + ii)) * 128;
    const float* v = qkv + ((size_t)(b * 128)) * 6144 + 4096 + h * 256 + d;
    float acc = 0.f;
    for (int j = 0; j < 128; j++) acc += srow[j] * v[(size_t)j * 6144];
    size_t o = ((size_t)(b * 128 + ii)) * 2048 + h * 256 + d;
    outF[o] = acc;
    outH[o] = (half_t)acc;
}

// farthest point sampling: one block per batch, all state in LDS
__global__ __launch_bounds__(512)
void k_fps(const float* __restrict__ x, int* __restrict__ out)
{
    int b = blockIdx.x, tid = threadIdx.x;     // 512 threads
    __shared__ float xs[3 * 2048];
    __shared__ float dmin[2048];
    __shared__ float bval[512];
    __shared__ int   bidx[512];
    __shared__ int   last;
    const float* xb = x + (size_t)b * 3 * 2048;
    for (int t = tid; t < 3 * 2048; t += 512) xs[t] = xb[t];
    for (int t = tid; t < 2048; t += 512) dmin[t] = 1e10f;
    if (tid == 0) { out[b * 1024] = 0; last = 0; }
    __syncthreads();
    for (int m = 1; m < 1024; m++) {
        int li = last;
        float lx = xs[li], ly = xs[2048 + li], lz = xs[4096 + li];
        float best = -1.f; int besti = 0;
        for (int p = tid; p < 2048; p += 512) {
            float dx = xs[p] - lx, dy = xs[2048 + p] - ly, dz = xs[4096 + p] - lz;
            float d = dx * dx + dy * dy + dz * dz;
            float dm = fminf(dmin[p], d);
            dmin[p] = dm;
            if (dm > best) { best = dm; besti = p; }   // strict > keeps lowest index
        }
        bval[tid] = best; bidx[tid] = besti;
        __syncthreads();
        for (int o = 256; o > 0; o >>= 1) {
            if (tid < o) {
                float v2 = bval[tid + o]; int i2 = bidx[tid + o];
                if (v2 > bval[tid] || (v2 == bval[tid] && i2 < bidx[tid])) { bval[tid] = v2; bidx[tid] = i2; }
            }
            __syncthreads();
        }
        if (tid == 0) { last = bidx[0]; out[b * 1024 + m] = bidx[0]; }
        __syncthreads();
    }
}

// gather: out = [x[:, :, idx] (4,3,1024) ; y[:, :, idx] (4,128,1024)]
__global__ void k_gather(const float* __restrict__ x, const float* __restrict__ y,
                         const int* __restrict__ idx, float* __restrict__ out)
{
    int i = blockIdx.x * blockDim.x + threadIdx.x;
    const int T1 = 4 * 3 * 1024;
    const int T2 = 4 * 128 * 1024;
    if (i >= T1 + T2) return;
    if (i < T1) {
        int m = i & 1023, c = (i >> 10) % 3, b = i / (3 * 1024);
        int id = idx[b * 1024 + m];
        out[i] = x[((size_t)(b * 3 + c)) * 2048 + id];
    } else {
        int j = i - T1;
        int m = j & 1023, c = (j >> 10) & 127, b = j >> 17;
        int id = idx[b * 1024 + m];
        out[i] = y[((size_t)(b * 128 + c)) * 2048 + id];
    }
}

// ---------------------------------------------------------------------------
static void launch_gemm(const half_t* A, const half_t* W, const float* bias,
                        const float* resid, float* outF, half_t* outH,
                        int M, int N, int K, int act, hipStream_t stream)
{
    int tiles  = (M >> 4) * (N >> 4);
    int blocks = (tiles + 7) / 8;           // 8 waves (256 thr) per block, wave32
    k_gemm_wmma<<<blocks, 256, 0, stream>>>(A, W, bias, resid, outF, outH, M, N, K, act);
}

extern "C" void kernel_launch(void* const* d_in, const int* in_sizes, int n_in,
                              void* d_out, int out_size, void* d_ws, size_t ws_size,
                              hipStream_t stream)
{
    const float* X     = (const float*)d_in[0];
    const float* W_IT  = (const float*)d_in[1];
    const float* B_IT  = (const float*)d_in[2];
    const float* WL1   = (const float*)d_in[3];
    const float* GN1G  = (const float*)d_in[4];
    const float* GN1B  = (const float*)d_in[5];
    const float* WL2   = (const float*)d_in[6];
    const float* GN2G  = (const float*)d_in[7];
    const float* GN2B  = (const float*)d_in[8];
    const float* E1N1G = (const float*)d_in[9];
    const float* E1N1B = (const float*)d_in[10];
    const float* E1QKV = (const float*)d_in[11];
    const float* E1PW  = (const float*)d_in[12];
    const float* E1PB  = (const float*)d_in[13];
    const float* E1MW  = (const float*)d_in[14];
    const float* E1MB  = (const float*)d_in[15];
    const float* E1C1  = (const float*)d_in[16];
    const float* E1C2  = (const float*)d_in[17];
    const float* E1N2G = (const float*)d_in[18];
    const float* E1N2B = (const float*)d_in[19];
    const float* E1F1W = (const float*)d_in[20];
    const float* E1F1B = (const float*)d_in[21];
    const float* E1F2W = (const float*)d_in[22];
    const float* E1F2B = (const float*)d_in[23];
    const float* E2N1G = (const float*)d_in[24];
    const float* E2N1B = (const float*)d_in[25];
    const float* E2QKV = (const float*)d_in[26];
    const float* E2PW  = (const float*)d_in[27];
    const float* E2PB  = (const float*)d_in[28];
    const float* E2N2G = (const float*)d_in[29];
    const float* E2N2B = (const float*)d_in[30];
    const float* E2F1W = (const float*)d_in[31];
    const float* E2F1B = (const float*)d_in[32];
    const float* E2F2W = (const float*)d_in[33];
    const float* E2F2B = (const float*)d_in[34];
    (void)in_sizes; (void)n_in; (void)out_size; (void)ws_size;

    char* base = (char*)d_ws; size_t off = 0;
    auto alloc = [&](size_t bytes) -> void* {
        off = (off + 255) & ~(size_t)255;
        void* p = base + off; off += bytes; return p;
    };

    float* f      = (float*)alloc((size_t)4 * 32 * 2048 * 4);
    int*   knnidx = (int*)  alloc((size_t)4 * 2048 * 16 * 4);
    float* part   = (float*)alloc((size_t)32 * 2048 * 4);
    float* stats  = (float*)alloc(64 * 4);
    float* mr     = (float*)alloc(64 * 4);
    float* knnf   = (float*)alloc((size_t)512 * 2048 * 4);
    float* nx     = (float*)alloc((size_t)12 * 2048 * 4);
    float* qkv1   = (float*)alloc((size_t)12 * 6144 * 4);
    float* ao1    = (float*)alloc((size_t)12 * 2048 * 4);
    float* x1     = (float*)alloc((size_t)12 * 2048 * 4);
    float* h1a    = (float*)alloc((size_t)4 * 512 * 2048 * 4);
    float* h1     = (float*)alloc((size_t)512 * 2048 * 4);
    float* y      = (float*)alloc((size_t)512 * 2048 * 4);
    float* qkv2   = (float*)alloc((size_t)512 * 6144 * 4);
    float* S      = (float*)alloc((size_t)4 * 8 * 128 * 128 * 4);
    float* aout   = (float*)alloc((size_t)512 * 2048 * 4);
    int*   fpsidx = (int*)  alloc((size_t)4 * 1024 * 4);
    half_t* am16  = (half_t*)alloc((size_t)512 * 4096 * 2);
    half_t* t16   = (half_t*)alloc((size_t)512 * 2048 * 2);
    half_t* g16   = (half_t*)alloc((size_t)512 * 4096 * 2);
    half_t* a16   = (half_t*)alloc((size_t)512 * 2048 * 2);
    half_t* wA    = (half_t*)alloc((size_t)6144 * 2048 * 2);   // big weight slot
    half_t* wB    = (half_t*)alloc((size_t)2048 * 4096 * 2);   // second weight slot

    // ---- edge-conv branch -------------------------------------------------
    k_intrans<<<(4 * 32 * 2048) / 256, 256, 0, stream>>>(X, W_IT, B_IT, f);
    k_knn<<<(4 * 2048) / 256, 256, 0, stream>>>(X, knnidx);
    k_edgeconv<<<8192, 256, 0, stream>>>(f, knnidx, WL1, WL2, GN1G, GN1B, GN2G, GN2B, mr, part, knnf, 0);
    k_redstats<<<32, 256, 0, stream>>>(part, stats, 0);
    k_gnfin<<<1, 16, 0, stream>>>(stats, mr, 0);
    k_edgeconv<<<8192, 256, 0, stream>>>(f, knnidx, WL1, WL2, GN1G, GN1B, GN2G, GN2B, mr, part, knnf, 1);
    k_redstats<<<32, 256, 0, stream>>>(part, stats, 32);
    k_gnfin<<<1, 16, 0, stream>>>(stats, mr, 1);
    k_edgeconv<<<8192, 256, 0, stream>>>(f, knnidx, WL1, WL2, GN1G, GN1B, GN2G, GN2B, mr, part, knnf, 2);

    // ---- attn over 3 tokens + 1x1 convs -----------------------------------
    k_ln<<<12, 256, 0, stream>>>(X, E1N1G, E1N1B, nx, (half_t*)nullptr, 2048);
    k_gemm_scalar<<<(12 * 6144 + 255) / 256, 256, 0, stream>>>(nx, E1QKV, nullptr, qkv1, 12, 6144, 2048);
    k_attn1<<<32, 256, 0, stream>>>(nx, qkv1, ao1);
    k_gemm_scalar<<<(12 * 2048 + 255) / 256, 256, 0, stream>>>(ao1, E1PW, E1PB, x1, 12, 2048, 2048);
    k_h1a<<<(4 * 512 * 2048) / 256, 256, 0, stream>>>(x1, E1C1, h1a);
    k_h1<<<(4 * 128 * 2048) / 256, 256, 0, stream>>>(h1a, E1C2, h1);

    // ---- merged GEMM: y = h1 + ([h1|knn_f] @ e1_mw.T + e1_mb) -------------
    k_pack<<<(512 * 4096) / 256, 256, 0, stream>>>(h1, knnf, am16);
    k_cvt<<<(2048 * 4096 + 255) / 256, 256, 0, stream>>>(E1MW, wB, 2048 * 4096);
    launch_gemm(am16, wB, E1MB, h1, y, nullptr, 512, 2048, 4096, 0, stream);

    // ---- MLP 1 ------------------------------------------------------------
    k_ln<<<512, 256, 0, stream>>>(y, E1N2G, E1N2B, nullptr, t16, 2048);
    k_cvt<<<(4096 * 2048 + 255) / 256, 256, 0, stream>>>(E1F1W, wA, 4096 * 2048);
    launch_gemm(t16, wA, E1F1B, nullptr, nullptr, g16, 512, 4096, 2048, 1, stream);
    k_cvt<<<(2048 * 4096 + 255) / 256, 256, 0, stream>>>(E1F2W, wB, 2048 * 4096);
    launch_gemm(g16, wB, E1F2B, y, y, nullptr, 512, 2048, 4096, 0, stream);

    // ---- attention 2 (seq=128, C=2048, 8 heads) ---------------------------
    k_ln<<<512, 256, 0, stream>>>(y, E2N1G, E2N1B, nullptr, t16, 2048);
    k_cvt<<<(6144 * 2048 + 255) / 256, 256, 0, stream>>>(E2QKV, wA, 6144 * 2048);
    launch_gemm(t16, wA, nullptr, nullptr, qkv2, nullptr, 512, 6144, 2048, 0, stream);
    k_scores<<<(4 * 8 * 128 * 128) / 256, 256, 0, stream>>>(qkv2, S);
    k_softmax128<<<4 * 8 * 128, 128, 0, stream>>>(S);
    k_attnO<<<(4 * 8 * 128 * 256) / 256, 256, 0, stream>>>(S, qkv2, aout, a16);
    k_cvt<<<(2048 * 2048 + 255) / 256, 256, 0, stream>>>(E2PW, wB, 2048 * 2048);
    launch_gemm(a16, wB, E2PB, y, y, nullptr, 512, 2048, 2048, 0, stream);

    // ---- MLP 2 ------------------------------------------------------------
    k_ln<<<512, 256, 0, stream>>>(y, E2N2G, E2N2B, nullptr, t16, 2048);
    k_cvt<<<(4096 * 2048 + 255) / 256, 256, 0, stream>>>(E2F1W, wA, 4096 * 2048);
    launch_gemm(t16, wA, E2F1B, nullptr, nullptr, g16, 512, 4096, 2048, 1, stream);
    k_cvt<<<(2048 * 4096 + 255) / 256, 256, 0, stream>>>(E2F2W, wB, 2048 * 4096);
    launch_gemm(g16, wB, E2F2B, y, y, nullptr, 512, 2048, 4096, 0, stream);

    // ---- FPS + gather -----------------------------------------------------
    k_fps<<<4, 512, 0, stream>>>(X, fpsidx);
    k_gather<<<(4 * 131 * 1024 + 255) / 256, 256, 0, stream>>>(X, y, fpsidx, (float*)d_out);
}
